// SidRewardHSTUBlock_35175782154741
// MI455X (gfx1250) — matrix operand order repaired and checked by reading, not compiled, not run on hardware
//
#include <hip/hip_runtime.h>
#include <math.h>
#include <stdint.h>

// ---------------- CDNA5 WMMA / TDM types ----------------
typedef __attribute__((ext_vector_type(16))) __bf16 v16bf;
typedef __attribute__((ext_vector_type(8)))  float  v8f;
typedef __attribute__((ext_vector_type(4)))  unsigned int u32x4;
typedef __attribute__((ext_vector_type(8)))  int          i32x8;
typedef __attribute__((ext_vector_type(4)))  int          i32x4;

static constexpr int kHid   = 512;
static constexpr int kHeads = 8;
static constexpr int kHd    = 64;
static constexpr int kSeq   = 2048;
static constexpr int kBatch = 4;
static constexpr int kRows  = kBatch * kSeq;   // 8192
static constexpr int kMaxSeq = 2048;

// ---------------- helpers ----------------
static __device__ __forceinline__ unsigned short f2bf(float f) {
  union { float f; unsigned u; } v; v.f = f;
  unsigned r = v.u + 0x7FFFu + ((v.u >> 16) & 1u);   // round-to-nearest-even
  return (unsigned short)(r >> 16);
}
static __device__ __forceinline__ float silu_f(float x) { return x / (1.0f + __expf(-x)); }

union FragBF { v16bf v; unsigned short s[16]; uint4 q[2]; };

// A tile 16x32 (MxK), row-major, ld elements per row.
static __device__ __forceinline__ v16bf load_frag_a(const unsigned short* base, int ld, int lane) {
  int m = lane & 15, kh = lane >> 4;
  const unsigned short* r = base + m * ld + kh * 8;
  FragBF u;
  u.q[0] = *(const uint4*)(r);        // K = 8*kh + 0..7
  u.q[1] = *(const uint4*)(r + 16);   // K = 8*kh + 16..23
  return u.v;
}

// B tile 32x16 (KxN) sourced from B^T stored row-major [N][K].
static __device__ __forceinline__ v16bf load_frag_bt(const unsigned short* bt, int ld, int lane) {
  int n = lane & 15, kh = lane >> 4;
  const unsigned short* r = bt + n * ld + kh * 8;
  FragBF u;
  u.q[0] = *(const uint4*)(r);
  u.q[1] = *(const uint4*)(r + 16);
  return u.v;
}

static __device__ __forceinline__ v8f wmma_bf16(v16bf a, v16bf b, v8f c) {
  return __builtin_amdgcn_wmma_f32_16x16x32_bf16(false, a, false, b, (short)0, c, false, false);
}

// ---------------- TDM: 2D tile (elements are 2 bytes) global -> LDS ----------------
// tile_k = dim0 (contiguous) length, tile_n = dim1 (rows); row_stride in elements.
static __device__ __forceinline__ void tdm_load_2d(unsigned lds_addr, const void* gptr,
                                                   int tile_k, int tile_n,
                                                   int tensor_k, int tensor_n,
                                                   int row_stride) {
  unsigned long long ga = (unsigned long long)(uintptr_t)gptr;
  u32x4 g0;
  g0[0] = 1u;                                               // count=1 (valid user D#)
  g0[1] = lds_addr;                                         // lds_addr [63:32]
  g0[2] = (unsigned)(ga & 0xFFFFFFFFu);                     // global_addr lo
  g0[3] = (unsigned)((ga >> 32) & 0x1FFFFFFu) | (2u << 30); // global_addr[56:32] | type=2
  i32x8 g1;
  g1[0] = (1 << 16);                                        // data_size=1 -> 2 bytes
  g1[1] = (int)(((unsigned)tensor_k & 0xFFFFu) << 16);      // tensor_dim0[15:0] @ bit48
  g1[2] = (int)((((unsigned)tensor_k >> 16) & 0xFFFFu) |
                (((unsigned)tensor_n & 0xFFFFu) << 16));    // dim0 hi | dim1 lo
  g1[3] = (int)((((unsigned)tensor_n >> 16) & 0xFFFFu) |
                (((unsigned)tile_k & 0xFFFFu) << 16));      // dim1 hi | tile_dim0
  g1[4] = (int)((unsigned)tile_n & 0xFFFFu);                // tile_dim1 (tile_dim2=0)
  g1[5] = row_stride;                                       // tensor_dim0_stride lo
  g1[6] = 0;
  g1[7] = 0;
  i32x4 z4 = {0, 0, 0, 0};
#if defined(__clang_major__) && (__clang_major__ >= 23)
  i32x8 z8 = {0, 0, 0, 0, 0, 0, 0, 0};
  __builtin_amdgcn_tensor_load_to_lds(g0, g1, z4, z4, z8, 0);
#else
  __builtin_amdgcn_tensor_load_to_lds(g0, g1, z4, z4, 0);
#endif
}

// ---------------- conversion kernels ----------------
__global__ void k_cvt_bf16(const float* __restrict__ in, unsigned short* __restrict__ out, int n) {
  int i = blockIdx.x * blockDim.x + threadIdx.x;
  if (i < n) out[i] = f2bf(in[i]);
}

// W[k][n] (512x512) -> Wt[n][k] bf16
__global__ void k_transpose_w(const float* __restrict__ w, unsigned short* __restrict__ wt) {
  int i = blockIdx.x * blockDim.x + threadIdx.x;
  int k = i >> 9, n = i & 511;
  wt[n * kHid + k] = f2bf(w[i]);
}

// ---------------- unified GEMM: act(X @ W + b) ----------------
// grid (kRows/64, kHid/256), block 256 (8 waves); wave -> 32x64 output tile.
// Weight tile (256 cols x 32 k, 16KB) staged to LDS via TDM, double-buffered.
// mode 0: Q bf16; 1: K bf16; 2: V bf16 transposed [b,h,d,s]; 3: U f32 (all silu);
// mode 4: f32 out, no activation (final projection).
__global__ void k_gemm(const unsigned short* __restrict__ X,
                       const unsigned short* __restrict__ Wt,
                       const float* __restrict__ bias,
                       unsigned short* __restrict__ outb,
                       float* __restrict__ outf,
                       int mode) {
  __shared__ __align__(16) unsigned short sB[2][256 * 32];   // [buf][n_local*32 + k_local]
  int lane = threadIdx.x & 31, wave = threadIdx.x >> 5;
  int mw = wave >> 2, nw = wave & 3;
  int mrow0 = blockIdx.x * 64 + mw * 32;     // this wave's first output row
  int ncol0 = blockIdx.y * 256;              // block's N base
  const unsigned short* wt_blk = Wt + ncol0 * kHid;

  if (wave == 0) {
    tdm_load_2d((unsigned)(uintptr_t)&sB[0][0], wt_blk, 32, 256, kHid, kHid, kHid);
    __builtin_amdgcn_s_wait_tensorcnt(0);
  }
  __syncthreads();

  v8f acc[2][4] = {};
  for (int kb = 0; kb < kHid; kb += 32) {
    int buf = (kb >> 5) & 1;
    if (wave == 0 && kb + 32 < kHid)
      tdm_load_2d((unsigned)(uintptr_t)&sB[buf ^ 1][0], wt_blk + (kb + 32),
                  32, 256, kHid, kHid, kHid);
    v16bf a[2];
#pragma unroll
    for (int i = 0; i < 2; ++i)
      a[i] = load_frag_a(X + (mrow0 + i * 16) * kHid + kb, kHid, lane);
#pragma unroll
    for (int j = 0; j < 4; ++j) {
      v16bf b = load_frag_bt(&sB[buf][(nw * 64 + j * 16) * 32], 32, lane);
      acc[0][j] = wmma_bf16(a[0], b, acc[0][j]);
      acc[1][j] = wmma_bf16(a[1], b, acc[1][j]);
    }
    if (wave == 0) __builtin_amdgcn_s_wait_tensorcnt(0);  // next buffer landed
    __syncthreads();                                      // + everyone done reading buf
  }

  int kh = lane >> 4, nloc = lane & 15;
#pragma unroll
  for (int i = 0; i < 2; ++i) {
#pragma unroll
    for (int j = 0; j < 4; ++j) {
      int n = ncol0 + nw * 64 + j * 16 + nloc;
      float bia = bias[n];
#pragma unroll
      for (int jj = 0; jj < 8; ++jj) {
        int m = mrow0 + i * 16 + jj + kh * 8;
        float v = acc[i][j][jj] + bia;
        if (mode == 4) {
          outf[m * kHid + n] = v;
        } else {
          v = silu_f(v);
          if (mode == 3) {
            outf[m * kHid + n] = v;
          } else if (mode == 2) {
            int bb = m >> 11, sq = m & (kSeq - 1), h = n >> 6, d = n & (kHd - 1);
            outb[((bb * kHeads + h) * kHd + d) * kSeq + sq] = f2bf(v);
          } else {
            outb[m * kHid + n] = f2bf(v);
          }
        }
      }
    }
  }
}

// ---------------- attention: AU = (silu(QK^T*s + relbias) . causal) @ V .* U ----------------
// grid (B*H, kSeq/(32*8)), block 256; wave -> 32 q rows, streams kv in 32-chunks.
__global__ void k_attn(const unsigned short* __restrict__ Qb,
                       const unsigned short* __restrict__ Kb,
                       const unsigned short* __restrict__ Vt,
                       const float* __restrict__ Uf,
                       const float* __restrict__ rel,
                       unsigned short* __restrict__ AU) {
  __shared__ __align__(16) unsigned short sS[8][32 * 32];   // per-wave 32x32 score tile
  int lane = threadIdx.x & 31, wave = threadIdx.x >> 5;
  int bh = blockIdx.x; int bb = bh >> 3, h = bh & 7;
  int q0 = (blockIdx.y * 8 + wave) * 32;
  int kh = lane >> 4, nloc = lane & 15;
  const float scale = 0.125f;  // HEAD_DIM^-0.5

  v16bf aq[2][2];
#pragma unroll
  for (int i = 0; i < 2; ++i)
#pragma unroll
    for (int c = 0; c < 2; ++c)
      aq[i][c] = load_frag_a(Qb + (bb * kSeq + q0 + i * 16) * kHid + h * kHd + c * 32, kHid, lane);

  v8f acc_o[2][4] = {};
  unsigned short* smem = sS[wave];

  for (int kv0 = 0; kv0 <= q0 + 31; kv0 += 32) {
#pragma unroll
    for (int t = 0; t < 2; ++t) {
      int kvb = kv0 + t * 16;
      v8f accs0 = {}, accs1 = {};
#pragma unroll
      for (int c = 0; c < 2; ++c) {
        v16bf bk = load_frag_bt(Kb + (bb * kSeq + kvb) * kHid + h * kHd + c * 32, kHid, lane);
        accs0 = wmma_bf16(aq[0][c], bk, accs0);   // K-frag reused for both q-tiles
        accs1 = wmma_bf16(aq[1][c], bk, accs1);
      }
      int n = kvb + nloc;
#pragma unroll
      for (int jj = 0; jj < 8; ++jj) {
        int m0 = q0 + jj + kh * 8;
        int m1 = m0 + 16;
        float s0 = accs0[jj] * scale + rel[(m0 - n + (kMaxSeq - 1)) * kHeads + h];
        float s1 = accs1[jj] * scale + rel[(m1 - n + (kMaxSeq - 1)) * kHeads + h];
        float w0 = (n <= m0) ? silu_f(s0) : 0.0f;
        float w1 = (n <= m1) ? silu_f(s1) : 0.0f;
        smem[(jj + kh * 8) * 32 + t * 16 + nloc] = f2bf(w0);
        smem[(16 + jj + kh * 8) * 32 + t * 16 + nloc] = f2bf(w1);
      }
    }
    // cross-lane C->A relayout through LDS; intra-wave only (DS in-order per wave)
    asm volatile("s_wait_dscnt 0" ::: "memory");
    v16bf as0 = load_frag_a(smem, 32, lane);
    v16bf as1 = load_frag_a(smem + 16 * 32, 32, lane);
#pragma unroll
    for (int nt4 = 0; nt4 < 4; ++nt4) {
      v16bf vb = load_frag_bt(Vt + ((bb * kHeads + h) * kHd + nt4 * 16) * kSeq + kv0, kSeq, lane);
      acc_o[0][nt4] = wmma_bf16(as0, vb, acc_o[0][nt4]);  // V-frag reused for both q-tiles
      acc_o[1][nt4] = wmma_bf16(as1, vb, acc_o[1][nt4]);
    }
  }

#pragma unroll
  for (int i = 0; i < 2; ++i)
#pragma unroll
    for (int nt4 = 0; nt4 < 4; ++nt4)
#pragma unroll
      for (int jj = 0; jj < 8; ++jj) {
        int m = q0 + i * 16 + jj + kh * 8;
        int col = h * kHd + nt4 * 16 + nloc;
        int idx = (bb * kSeq + m) * kHid + col;
        AU[idx] = f2bf(acc_o[i][nt4][jj] * Uf[idx]);
      }
}

// ---------------- launcher ----------------
extern "C" void kernel_launch(void* const* d_in, const int* in_sizes, int n_in,
                              void* d_out, int out_size, void* d_ws, size_t ws_size,
                              hipStream_t stream) {
  (void)in_sizes; (void)n_in; (void)out_size; (void)ws_size;
  const float* query = (const float*)d_in[0];
  const float* key_  = (const float*)d_in[1];
  const float* value = (const float*)d_in[2];
  // d_in[3] = attn_mask (causal tril) -> applied analytically in k_attn
  const float* Wq = (const float*)d_in[4];
  const float* bq = (const float*)d_in[5];
  const float* Wk = (const float*)d_in[6];
  const float* bk = (const float*)d_in[7];
  const float* Wv = (const float*)d_in[8];
  const float* bv = (const float*)d_in[9];
  const float* Wu = (const float*)d_in[10];
  const float* bu = (const float*)d_in[11];
  const float* Wf = (const float*)d_in[12];
  const float* bf = (const float*)d_in[13];
  const float* rel = (const float*)d_in[14];
  float* out = (float*)d_out;

  char* ws = (char*)d_ws;
  size_t off = 0;
  auto alloc = [&](size_t bytes) -> void* {
    void* p = ws + off;
    off += (bytes + 255) & ~size_t(255);
    return p;
  };
  const size_t actB = (size_t)kRows * kHid;  // 8192*512
  unsigned short* Xq  = (unsigned short*)alloc(actB * 2);
  unsigned short* Xk  = (unsigned short*)alloc(actB * 2);
  unsigned short* Xv  = (unsigned short*)alloc(actB * 2);
  unsigned short* Wqt = (unsigned short*)alloc((size_t)kHid * kHid * 2);
  unsigned short* Wkt = (unsigned short*)alloc((size_t)kHid * kHid * 2);
  unsigned short* Wvt = (unsigned short*)alloc((size_t)kHid * kHid * 2);
  unsigned short* Wut = (unsigned short*)alloc((size_t)kHid * kHid * 2);
  unsigned short* Wft = (unsigned short*)alloc((size_t)kHid * kHid * 2);
  unsigned short* Qb  = (unsigned short*)alloc(actB * 2);
  unsigned short* Kb  = (unsigned short*)alloc(actB * 2);
  unsigned short* Vt  = (unsigned short*)alloc(actB * 2);   // [b,h,d,s]
  float*          Uf  = (float*)alloc(actB * 4);
  unsigned short* AU  = (unsigned short*)alloc(actB * 2);

  const int nAct = (int)actB;           // 4194304
  const int nW   = kHid * kHid;         // 262144
  k_cvt_bf16<<<(nAct + 255) / 256, 256, 0, stream>>>(query, Xq, nAct);
  k_cvt_bf16<<<(nAct + 255) / 256, 256, 0, stream>>>(key_,  Xk, nAct);
  k_cvt_bf16<<<(nAct + 255) / 256, 256, 0, stream>>>(value, Xv, nAct);
  k_transpose_w<<<(nW + 255) / 256, 256, 0, stream>>>(Wq, Wqt);
  k_transpose_w<<<(nW + 255) / 256, 256, 0, stream>>>(Wk, Wkt);
  k_transpose_w<<<(nW + 255) / 256, 256, 0, stream>>>(Wv, Wvt);
  k_transpose_w<<<(nW + 255) / 256, 256, 0, stream>>>(Wu, Wut);
  k_transpose_w<<<(nW + 255) / 256, 256, 0, stream>>>(Wf, Wft);

  dim3 pg(kRows / 64, kHid / 256);      // (128, 2), 8 waves/block, 32x64 per wave
  k_gemm<<<pg, 256, 0, stream>>>(Xq, Wqt, bq, Qb, nullptr, 0);
  k_gemm<<<pg, 256, 0, stream>>>(Xk, Wkt, bk, Kb, nullptr, 1);
  k_gemm<<<pg, 256, 0, stream>>>(Xv, Wvt, bv, Vt, nullptr, 2);
  k_gemm<<<pg, 256, 0, stream>>>(Xq, Wut, bu, nullptr, Uf, 3);

  dim3 ag(kBatch * kHeads, kSeq / 256); // (32, 8), 8 waves/block, 32 q-rows per wave
  k_attn<<<ag, 256, 0, stream>>>(Qb, Kb, Vt, Uf, rel, AU);

  k_gemm<<<pg, 256, 0, stream>>>(AU, Wft, bf, nullptr, out, 4);
}
